// CCSA_80315888435505
// MI455X (gfx1250) — matrix operand order repaired and checked by reading, not compiled, not run on hardware
//
#include <hip/hip_runtime.h>
#include <hip/hip_bf16.h>

typedef __attribute__((ext_vector_type(16))) __bf16 v16bf;
typedef __attribute__((ext_vector_type(8)))  __bf16 v8bf;
typedef __attribute__((ext_vector_type(8)))  float  v8f;

#define NS 8192
#define NT 8192
#define DIM 512
#define MARGIN 0.5f

// ---------------------------------------------------------------------------
// Kernel 0: zero the output (we accumulate with atomics).
// ---------------------------------------------------------------------------
__global__ void ccsa_zero_out(float* __restrict__ out) {
    int i = blockIdx.x * blockDim.x + threadIdx.x;
    if (i < 2 * NS) out[i] = 0.0f;
}

// ---------------------------------------------------------------------------
// Kernel 1: f32 -> bf16 conversion (RNE) + row squared-norms.
// One wave (32 lanes) per row; 16384 rows total (source then target).
// ---------------------------------------------------------------------------
__device__ inline unsigned short f2bf_rne(float f) {
    unsigned int u = __float_as_uint(f);
    unsigned int r = u + 0x7FFFu + ((u >> 16) & 1u);
    return (unsigned short)(r >> 16);
}

__global__ void ccsa_prep(const float* __restrict__ src,
                          const float* __restrict__ tgt,
                          float* __restrict__ sq_s,
                          float* __restrict__ sq_t,
                          unsigned short* __restrict__ sbf,
                          unsigned short* __restrict__ tbf) {
    int wave = (blockIdx.x * blockDim.x + threadIdx.x) >> 5;
    int lane = threadIdx.x & 31;

    const float* inp;
    float* sq;
    unsigned short* ob;
    int row;
    if (wave < NS) { inp = src; sq = sq_s; ob = sbf; row = wave; }
    else           { inp = tgt; sq = sq_t; ob = tbf; row = wave - NS; }

    const float* rp = inp + (size_t)row * DIM;
    unsigned short* op = ob + (size_t)row * DIM;

    float s = 0.0f;
#pragma unroll
    for (int i = 0; i < DIM / 32; ++i) {
        float v = rp[lane + i * 32];
        s += v * v;
        op[lane + i * 32] = f2bf_rne(v);
    }
#pragma unroll
    for (int m = 16; m >= 1; m >>= 1) s += __shfl_xor(s, m, 32);
    if (lane == 0) sq[row] = s;
}

// ---------------------------------------------------------------------------
// Kernel 2: tiled Gram + fused loss reduction.
// Block = 256 threads (8 waves). M-tile = 128 source rows in LDS (bf16).
// Each wave owns 16 rows and sweeps 1024 target columns, 16 at a time,
// doing K=512 via 16x v_wmma_f32_16x16x32_bf16 per tile.
// ---------------------------------------------------------------------------
#define ROW_PAD 520   // 512 + 8 bf16 -> +16B row pad to break bank conflicts

__global__ __launch_bounds__(256, 1) void ccsa_main(
    const unsigned short* __restrict__ sbf,
    const unsigned short* __restrict__ tbf,
    const float* __restrict__ sq_s,
    const float* __restrict__ sq_t,
    const int* __restrict__ ssec,
    const int* __restrict__ tsec,
    float* __restrict__ out) {

    __shared__ unsigned short Atile[128][ROW_PAD];

    const int tid   = threadIdx.x;
    const int wave  = tid >> 5;
    const int lane  = tid & 31;
    const int mBlk  = blockIdx.x * 128;
    const int nBase = blockIdx.y * 1024;

    // ---- cooperative load of the 128x512 bf16 source tile into LDS ----
    {
        int row  = tid >> 1;        // 0..127
        int half = tid & 1;         // 0/1 -> 256-element halves
        const uint4* gp = (const uint4*)(sbf + (size_t)(mBlk + row) * DIM + half * 256);
        uint4* lp = (uint4*)(&Atile[row][half * 256]);
#pragma unroll
        for (int i = 0; i < 32; ++i) lp[i] = gp[i];
    }
    __syncthreads();

    const int laneHi = lane >> 4;     // half-wave select
    const int lane15 = lane & 15;
    const int mWave  = wave * 16;     // this wave's row offset inside tile

    // Per-lane source-row parameters: VGPR r of C holds row (mWave + 8*laneHi + r).
    float sqs_r[8];
    int   sec_r[8];
#pragma unroll
    for (int r = 0; r < 8; ++r) {
        int m = mBlk + mWave + laneHi * 8 + r;
        sqs_r[r] = sq_s[m];
        sec_r[r] = ssec[m];
    }

    float accS[8], accC[8];
#pragma unroll
    for (int r = 0; r < 8; ++r) { accS[r] = 0.0f; accC[r] = 0.0f; }

    // A-fragment addressing (contiguous K, matching B):
    //  lane<16 : row = mWave+lane,    K = k0..k0+15
    //  lane>=16: row = mWave+lane-16, K = k0+16..k0+31
    const unsigned short* aRow = &Atile[mWave + lane15][0];
    const int aOff = laneHi * 16;

    for (int n0 = nBase; n0 < nBase + 1024; n0 += 16) {
        const int n = n0 + lane15;
        // B-fragment (32x16): lane<16 holds col n, K=k0..k0+15 (contiguous);
        // lane>=16 holds col n, K=k0+16..k0+31.
        const unsigned short* bRow = tbf + (size_t)n * DIM + laneHi * 16;

        v8f c = {};
#pragma unroll
        for (int k0 = 0; k0 < DIM; k0 += 32) {
            v8bf alo = *(const v8bf*)(aRow + k0 + aOff);
            v8bf ahi = *(const v8bf*)(aRow + k0 + aOff + 8);
            v16bf a = __builtin_shufflevector(alo, ahi,
                        0, 1, 2, 3, 4, 5, 6, 7, 8, 9, 10, 11, 12, 13, 14, 15);
            v16bf b = *(const v16bf*)(bRow + k0);
            c = __builtin_amdgcn_wmma_f32_16x16x32_bf16(
                    false, a, false, b, (short)0, c, false, false);
        }

        // ---- fused epilogue: distance -> semantic / contrastive partials ----
        const float sqt_n = sq_t[n];
        const int   tsn   = tsec[n];
#pragma unroll
        for (int r = 0; r < 8; ++r) {
            float d2 = sqs_r[r] + sqt_n - 2.0f * c[r];
            d2 = fmaxf(d2, 0.0f) * (1.0f / (float)DIM);
            float d = sqrtf(d2);
            float h = fmaxf(MARGIN - d, 0.0f);
            bool same = (sec_r[r] == tsn);
            accS[r] += same ? d2 : 0.0f;
            accC[r] += same ? 0.0f : h * h;
        }
    }

    // ---- reduce across the 16 lanes of each half-wave (masks stay in-half) ----
#pragma unroll
    for (int r = 0; r < 8; ++r) {
        float s = accS[r], cc = accC[r];
#pragma unroll
        for (int m = 8; m >= 1; m >>= 1) {
            s  += __shfl_xor(s, m, 32);
            cc += __shfl_xor(cc, m, 32);
        }
        if (lane15 == 0) {
            int m = mBlk + mWave + laneHi * 8 + r;
            atomicAdd(&out[m],      s  * (1.0f / (float)NT));
            atomicAdd(&out[NS + m], cc * (1.0f / (float)NT));
        }
    }
}

// ---------------------------------------------------------------------------
extern "C" void kernel_launch(void* const* d_in, const int* in_sizes, int n_in,
                              void* d_out, int out_size, void* d_ws, size_t ws_size,
                              hipStream_t stream) {
    const float* src  = (const float*)d_in[0];
    const float* tgt  = (const float*)d_in[1];
    const int*   ssec = (const int*)d_in[2];
    const int*   tsec = (const int*)d_in[3];
    float* out = (float*)d_out;

    char* ws = (char*)d_ws;
    float* sq_s = (float*)(ws);
    float* sq_t = (float*)(ws + 32768);
    unsigned short* sbf = (unsigned short*)(ws + 65536);
    unsigned short* tbf = (unsigned short*)(ws + 65536 + (size_t)NS * DIM * 2);

    ccsa_zero_out<<<(2 * NS + 255) / 256, 256, 0, stream>>>(out);
    ccsa_prep<<<(NS + NT) / 8, 256, 0, stream>>>(src, tgt, sq_s, sq_t, sbf, tbf);

    dim3 grid(NS / 128, NT / 1024);
    ccsa_main<<<grid, 256, 0, stream>>>(sbf, tbf, sq_s, sq_t, ssec, tsec, out);
}
